// QueryEncoder_86131274154741
// MI455X (gfx1250) — matrix operand order repaired and checked by reading, not compiled, not run on hardware
//
#include <hip/hip_runtime.h>
#include <hip/hip_bf16.h>

typedef __bf16 bf16_t;
typedef __attribute__((ext_vector_type(16))) __bf16 v16bf;
typedef __attribute__((ext_vector_type(8)))  __bf16 v8bf;
typedef __attribute__((ext_vector_type(8)))  float  v8f;

#define B_Q   16384
#define D_IN  128
#define H_SZ  256
#define G4    1024
#define S_SUP 4096
#define S_SLAB (S_SUP / 4)   // support rows per wave in attn (4-way split)

// ---------------------------------------------------------------------------
// WMMA fragment loaders (bf16, 16x16x32). Layouts per CDNA5 ISA:
//  A (16x32, MxK): lanes 0-15 hold row M=lane, K = {0..7, 16..23};
//                  lanes 16-31 hold row M=lane-16, K = {8..15, 24..31}.
//  B (32x16, KxN): lane holds col N=lane&15; lanes 0-15: K = {0..7, 16..23},
//                  lanes 16-31: K = {8..15, 24..31} (same order as A).
// Both reduce to contiguous 16B/32B loads from row-major [rows, K] storage.
// ---------------------------------------------------------------------------
static __device__ __forceinline__ v16bf cat16(v8bf lo, v8bf hi) {
  return __builtin_shufflevector(lo, hi, 0,1,2,3,4,5,6,7,8,9,10,11,12,13,14,15);
}

static __device__ __forceinline__ v16bf load_a_frag(const bf16_t* base, int ld, int lane) {
  int k0 = (lane & 16) >> 1;                       // 0 or 8
  const bf16_t* p = base + (lane & 15) * ld + k0;  // row-major [M, K]
  v8bf lo = *reinterpret_cast<const v8bf*>(p);
  v8bf hi = *reinterpret_cast<const v8bf*>(p + 16);
  return cat16(lo, hi);
}

static __device__ __forceinline__ v16bf load_b_frag(const bf16_t* base, int ld, int lane) {
  int k0 = (lane & 16) >> 1;                       // 0 or 8
  const bf16_t* p = base + (lane & 15) * ld + k0;  // row-major [N, K]
  v8bf lo = *reinterpret_cast<const v8bf*>(p);
  v8bf hi = *reinterpret_cast<const v8bf*>(p + 16);
  return cat16(lo, hi);
}

static __device__ __forceinline__ v8f wmma_bf16(v16bf a, v16bf b, v8f c) {
  return __builtin_amdgcn_wmma_f32_16x16x32_bf16(false, a, false, b, (short)0, c, false, false);
}

static __device__ __forceinline__ float sigm(float x) {
  return 1.0f / (1.0f + __expf(-x));
}

// ---------------------------------------------------------------------------
// Setup kernels
// ---------------------------------------------------------------------------
__global__ void cvt_f32_bf16_kernel(const float* __restrict__ in, bf16_t* __restrict__ out, int n) {
  int i = blockIdx.x * blockDim.x + threadIdx.x;
  if (i < n) out[i] = (bf16_t)in[i];
}

__global__ void transpose_support_kernel(const float* __restrict__ s, bf16_t* __restrict__ st) {
  int i = blockIdx.x * blockDim.x + threadIdx.x;   // over S*D
  int row = i >> 7;                                 // support row (0..4095)
  int col = i & 127;                                // d (0..127)
  if (i < S_SUP * D_IN) st[col * S_SUP + row] = (bf16_t)s[i];
}

__global__ void bias_sum_kernel(const float* __restrict__ bi, const float* __restrict__ bh,
                                float* __restrict__ bs) {
  int i = blockIdx.x * blockDim.x + threadIdx.x;
  if (i < G4) bs[i] = bi[i] + bh[i];
}

__global__ void zero_bf16_kernel(bf16_t* p, int n) {
  int i = blockIdx.x * blockDim.x + threadIdx.x;
  if (i < n) p[i] = (bf16_t)0.0f;
}

__global__ void zero_f32_kernel(float* p, int n) {
  int i = blockIdx.x * blockDim.x + threadIdx.x;
  if (i < n) p[i] = 0.0f;
}

// ---------------------------------------------------------------------------
// x_gates = query @ W_ih.T + (b_ih + b_hh)       [B, 4H] f32
// One wave computes one 16x16 tile; K = 128 -> 4 WMMAs.
// grid (B/16, 4H/16/4), block 128 (4 waves).
// ---------------------------------------------------------------------------
__global__ void xgates_kernel(const bf16_t* __restrict__ q, const bf16_t* __restrict__ wih,
                              const float* __restrict__ bsum, float* __restrict__ xg) {
  int lane = threadIdx.x & 31;
  int wave = threadIdx.x >> 5;
  int rowb = blockIdx.x * 16;
  int col  = (blockIdx.y * 4 + wave) * 16;
  v8f acc = {};
#pragma unroll
  for (int k = 0; k < D_IN; k += 32) {
    v16bf a = load_a_frag(q + rowb * D_IN + k, D_IN, lane);
    v16bf b = load_b_frag(wih + col * D_IN + k, D_IN, lane);
    acc = wmma_bf16(a, b, acc);
  }
  int n = lane & 15, hb = lane >> 4;
  float bias = bsum[col + n];
#pragma unroll
  for (int r = 0; r < 8; ++r) {
    int row = rowb + r + hb * 8;
    xg[(size_t)row * G4 + col + n] = acc[r] + bias;
  }
}

// ---------------------------------------------------------------------------
// gates = x_gates + h_r @ W_hh.T; LSTM cell; h = query + h_full[:, :D].
// One wave owns a 16-row x 16-col slice of h' in [0,256) and computes all
// four gate quadrants (same (row, h') index) -> 4 accumulators, K=256.
// grid (B/16, 4), block 128 (4 waves; wave w -> col slice blockIdx.y*4+w).
// ---------------------------------------------------------------------------
__global__ void cell_kernel(const bf16_t* __restrict__ hr_in, const bf16_t* __restrict__ whh,
                            const float* __restrict__ xg, const float* __restrict__ q32,
                            float* __restrict__ c_st, bf16_t* __restrict__ hr_out,
                            float* __restrict__ outh, int write_out) {
  int lane = threadIdx.x & 31;
  int wave = threadIdx.x >> 5;
  int rowb = blockIdx.x * 16;
  int c0   = (blockIdx.y * 4 + wave) * 16;   // in [0, 256)
  v8f ai = {}, af = {}, ag = {}, ao = {};
#pragma unroll
  for (int k = 0; k < H_SZ; k += 32) {
    v16bf a = load_a_frag(hr_in + rowb * H_SZ + k, H_SZ, lane);
    ai = wmma_bf16(a, load_b_frag(whh + (c0 + 0 * H_SZ) * H_SZ + k, H_SZ, lane), ai);
    af = wmma_bf16(a, load_b_frag(whh + (c0 + 1 * H_SZ) * H_SZ + k, H_SZ, lane), af);
    ag = wmma_bf16(a, load_b_frag(whh + (c0 + 2 * H_SZ) * H_SZ + k, H_SZ, lane), ag);
    ao = wmma_bf16(a, load_b_frag(whh + (c0 + 3 * H_SZ) * H_SZ + k, H_SZ, lane), ao);
  }
  int n = lane & 15, hb = lane >> 4;
  int col = c0 + n;
#pragma unroll
  for (int r = 0; r < 8; ++r) {
    int row = rowb + r + hb * 8;
    const float* xrow = xg + (size_t)row * G4;
    float gi = ai[r] + xrow[col + 0 * H_SZ];
    float gf = af[r] + xrow[col + 1 * H_SZ];
    float gg = ag[r] + xrow[col + 2 * H_SZ];
    float go = ao[r] + xrow[col + 3 * H_SZ];
    float cn = sigm(gf) * c_st[row * H_SZ + col] + sigm(gi) * tanhf(gg);
    c_st[row * H_SZ + col] = cn;
    float hf = sigm(go) * tanhf(cn);
    if (col < D_IN) {
      float h = q32[row * D_IN + col] + hf;       // residual on first half
      hr_out[row * H_SZ + col] = (bf16_t)h;
      if (write_out) outh[row * D_IN + col] = h;
    }
  }
}

// ---------------------------------------------------------------------------
// Flash attention: r = softmax(h @ support.T) @ support -> hr[:, 128:256].
// Block = 128 threads (4 waves) owns 16 query rows; the S=4096 reduction is
// split 4-ways across the waves (1024 support rows each) for occupancy, and
// the partial flash states (m, l, weighted r) are merged through LDS.
// The softmax row-sum is computed by the matrix engine as an extra
// accumulator tile with an all-ones B fragment (l = P @ 1), which both
// removes the 32-shuffle sum tree per chunk and leaves the row sum
// replicated across lanes exactly as the merge phase needs it.
// grid (B/16), block 128.
// ---------------------------------------------------------------------------
__global__ void attn_kernel(const bf16_t* __restrict__ hr, const bf16_t* __restrict__ supp,
                            const bf16_t* __restrict__ suppT, bf16_t* __restrict__ hr_r) {
  __shared__ float  rbuf[4][16][128];   // per-wave partial numerators (32 KB)
  __shared__ float  mbuf[4][16];        // per-wave row maxima
  __shared__ float  lbuf[4][16];        // per-wave row sums (rescaled)
  __shared__ bf16_t plds[4][16 * 32];   // per-wave P staging (C->A transpose)

  int lane = threadIdx.x & 31;
  int wave = threadIdx.x >> 5;
  int rowb = blockIdx.x * 16;
  int n = lane & 15, hb = lane >> 4;
  bf16_t* pls = &plds[wave][0];

  // h A-fragments, resident for the whole S loop (K = 128 -> 4 frags)
  v16bf hA[4];
#pragma unroll
  for (int kk = 0; kk < 4; ++kk)
    hA[kk] = load_a_frag(hr + rowb * H_SZ + kk * 32, H_SZ, lane);

  // all-ones B fragment for the row-sum accumulator
  v16bf bones;
#pragma unroll
  for (int i = 0; i < 16; ++i) bones[i] = (bf16_t)1.0f;

  v8f zero = {};
  v8f acc[9];                           // [0..7] r numerator tiles, [8] row sum
#pragma unroll
  for (int t = 0; t < 9; ++t) acc[t] = zero;
  float mrun[8];
#pragma unroll
  for (int r = 0; r < 8; ++r) mrun[r] = -3.0e30f;

  int s0 = wave * S_SLAB;
  for (int s = s0; s < s0 + S_SLAB; s += 32) {
    __builtin_prefetch(supp + (s + 64) * D_IN, 0, 0);
    // logits for 32 support rows: two 16x16 tiles
    v8f cA = zero, cB = zero;
#pragma unroll
    for (int kk = 0; kk < 4; ++kk) {
      cA = wmma_bf16(hA[kk], load_b_frag(supp + (s + 0)  * D_IN + kk * 32, D_IN, lane), cA);
      cB = wmma_bf16(hA[kk], load_b_frag(supp + (s + 16) * D_IN + kk * 32, D_IN, lane), cB);
    }
    // online max per row (row m = r + hb*8 lives in one 16-lane half;
    // xor masks 1,2,4,8 stay within the half)
    float p0[8], p1[8];
#pragma unroll
    for (int r = 0; r < 8; ++r) {
      float mx = fmaxf(cA[r], cB[r]);
      mx = fmaxf(mx, __shfl_xor(mx, 1, 32));
      mx = fmaxf(mx, __shfl_xor(mx, 2, 32));
      mx = fmaxf(mx, __shfl_xor(mx, 4, 32));
      mx = fmaxf(mx, __shfl_xor(mx, 8, 32));
      float mnew = fmaxf(mrun[r], mx);
      float alpha = __expf(mrun[r] - mnew);
      mrun[r] = mnew;
#pragma unroll
      for (int t = 0; t < 9; ++t) acc[t][r] *= alpha;
      p0[r] = __expf(cA[r] - mnew);
      p1[r] = __expf(cB[r] - mnew);
    }
    // C-layout -> A-layout transpose of P (16x32) through per-wave LDS
#pragma unroll
    for (int r = 0; r < 8; ++r) {
      int m = r + hb * 8;
      pls[m * 32 + n]      = (bf16_t)p0[r];
      pls[m * 32 + 16 + n] = (bf16_t)p1[r];
    }
    asm volatile("s_wait_dscnt 0" ::: "memory");    // wave-local LDS RAW fence
    v16bf pA = load_a_frag(pls, 32, lane);
    // r += P @ support[s:s+32, :]; l += P @ 1
#pragma unroll
    for (int t = 0; t < 8; ++t)
      acc[t] = wmma_bf16(pA, load_b_frag(suppT + (t * 16) * S_SUP + s, S_SUP, lane), acc[t]);
    acc[8] = wmma_bf16(pA, bones, acc[8]);
  }

  // ---- merge the 4 per-wave partial flash states --------------------------
#pragma unroll
  for (int r = 0; r < 8; ++r) {
    int row = r + hb * 8;
    mbuf[wave][row] = mrun[r];          // duplicate same-value writes: benign
  }
  __syncthreads();
#pragma unroll
  for (int r = 0; r < 8; ++r) {
    int row = r + hb * 8;
    float g = fmaxf(fmaxf(mbuf[0][row], mbuf[1][row]),
                    fmaxf(mbuf[2][row], mbuf[3][row]));
    float sc = __expf(mrun[r] - g);
#pragma unroll
    for (int t = 0; t < 9; ++t) acc[t][r] *= sc;
  }
#pragma unroll
  for (int r = 0; r < 8; ++r) {
    int row = r + hb * 8;
    lbuf[wave][row] = acc[8][r];        // row sum is lane-replicated (P @ 1)
#pragma unroll
    for (int t = 0; t < 8; ++t)
      rbuf[wave][row][t * 16 + n] = acc[t][r];
  }
  __syncthreads();
  // final combine: 128 threads x 16 elements; each thread owns one row slice
  int tid = threadIdx.x;
  int row = tid >> 3;
  int col0 = (tid & 7) * 16;
  float den = lbuf[0][row] + lbuf[1][row] + lbuf[2][row] + lbuf[3][row];
  float inv = 1.0f / den;
#pragma unroll
  for (int j = 0; j < 16; ++j) {
    int col = col0 + j;
    float num = rbuf[0][row][col] + rbuf[1][row][col] +
                rbuf[2][row][col] + rbuf[3][row][col];
    hr_r[(rowb + row) * H_SZ + D_IN + col] = (bf16_t)(num * inv);
  }
}

// ---------------------------------------------------------------------------
extern "C" void kernel_launch(void* const* d_in, const int* in_sizes, int n_in,
                              void* d_out, int out_size, void* d_ws, size_t ws_size,
                              hipStream_t stream) {
  const float* support = (const float*)d_in[0];
  const float* query   = (const float*)d_in[1];
  const float* W_ih    = (const float*)d_in[2];
  const float* W_hh    = (const float*)d_in[3];
  const float* b_ih    = (const float*)d_in[4];
  const float* b_hh    = (const float*)d_in[5];
  float* out = (float*)d_out;

  char* p = (char*)d_ws;
  auto carve = [&](size_t bytes) -> char* {
    char* r = p;
    p += (bytes + 255) & ~(size_t)255;
    return r;
  };
  bf16_t* q_bf   = (bf16_t*)carve((size_t)B_Q * D_IN * 2);
  bf16_t* wih_bf = (bf16_t*)carve((size_t)G4 * D_IN * 2);
  bf16_t* whh_bf = (bf16_t*)carve((size_t)G4 * H_SZ * 2);
  bf16_t* sup_bf = (bf16_t*)carve((size_t)S_SUP * D_IN * 2);
  bf16_t* supT   = (bf16_t*)carve((size_t)D_IN * S_SUP * 2);
  float*  bsum   = (float*)carve((size_t)G4 * 4);
  float*  xg     = (float*)carve((size_t)B_Q * G4 * 4);
  bf16_t* hrA    = (bf16_t*)carve((size_t)B_Q * H_SZ * 2);
  bf16_t* hrB    = (bf16_t*)carve((size_t)B_Q * H_SZ * 2);
  float*  c_st   = (float*)carve((size_t)B_Q * H_SZ * 4);

  const int TPB = 256;
  cvt_f32_bf16_kernel<<<(B_Q * D_IN + TPB - 1) / TPB, TPB, 0, stream>>>(query, q_bf, B_Q * D_IN);
  cvt_f32_bf16_kernel<<<(G4 * D_IN + TPB - 1) / TPB, TPB, 0, stream>>>(W_ih, wih_bf, G4 * D_IN);
  cvt_f32_bf16_kernel<<<(G4 * H_SZ + TPB - 1) / TPB, TPB, 0, stream>>>(W_hh, whh_bf, G4 * H_SZ);
  cvt_f32_bf16_kernel<<<(S_SUP * D_IN + TPB - 1) / TPB, TPB, 0, stream>>>(support, sup_bf, S_SUP * D_IN);
  transpose_support_kernel<<<(S_SUP * D_IN + TPB - 1) / TPB, TPB, 0, stream>>>(support, supT);
  bias_sum_kernel<<<(G4 + TPB - 1) / TPB, TPB, 0, stream>>>(b_ih, b_hh, bsum);
  zero_bf16_kernel<<<(B_Q * H_SZ + TPB - 1) / TPB, TPB, 0, stream>>>(hrA, B_Q * H_SZ);
  zero_f32_kernel<<<(B_Q * H_SZ + TPB - 1) / TPB, TPB, 0, stream>>>(c_st, B_Q * H_SZ);

  xgates_kernel<<<dim3(B_Q / 16, G4 / 16 / 4), 128, 0, stream>>>(q_bf, wih_bf, bsum, xg);

  bf16_t* hin = hrA;
  bf16_t* hout = hrB;
  for (int step = 0; step < 4; ++step) {
    cell_kernel<<<dim3(B_Q / 16, 4), 128, 0, stream>>>(hin, whh_bf, xg, query, c_st, hout,
                                                       out, step == 3 ? 1 : 0);
    if (step < 3)   // last step's attention does not affect the output
      attn_kernel<<<dim3(B_Q / 16), 128, 0, stream>>>(hout, sup_bf, supT, hout);
    bf16_t* tmp = hin; hin = hout; hout = tmp;
  }
}
